// CentralVCriticHg_59700045414731
// MI455X (gfx1250) — compile-verified, matrix operand order checked by reading, not run on hardware
//
#include <hip/hip_runtime.h>

// ---------------------------------------------------------------------------
// Sizes from the reference
// ---------------------------------------------------------------------------
#define BS   32
#define T_   128
#define A_   16
#define SD   256
#define OD   128
#define F_   64
#define E_   8
#define EMB  64
#define HID  256
#define R_   256
#define EPS  1e-6f

#define K1PAD 352              // SD + A + EMB = 336 padded to 11*32
#define NROWS (BS * T_ * A_)   // 65536 MLP rows
#define BT    (BS * T_)        // 4096 GRU rows

typedef _Float16 half_t;
typedef __attribute__((ext_vector_type(16))) _Float16 v16h;
typedef __attribute__((ext_vector_type(8)))  _Float16 v8h;
typedef __attribute__((ext_vector_type(8)))  float    v8f;
typedef __attribute__((ext_vector_type(4)))  float    f32x4;
typedef __attribute__((ext_vector_type(4)))  unsigned int u32x4;
typedef __attribute__((ext_vector_type(8)))  int      i32x8;
typedef __attribute__((ext_vector_type(4)))  int      i32x4;

#if __has_builtin(__builtin_amdgcn_tensor_load_to_lds) && __has_builtin(__builtin_amdgcn_s_wait_tensorcnt)
#define USE_TDM 1
#else
#define USE_TDM 0
#endif

// ---------------------------------------------------------------------------
// WMMA helpers (v_wmma_f32_16x16x32_f16, D = A(16x32)*B(32x16)+C)
// f16 fragment K-pattern per ISA 7.12.2: element i -> K = i + (i>=8?8:0) + kb,
// kb = (lane>=16 ? 8 : 0).  So a fragment is TWO CONTIGUOUS 8-half chunks at
// [kb, kb+8) and [kb+16, kb+24): loadable as two b128 ops when the matrix is
// laid out with K contiguous (A row-major; B stored TRANSPOSED as NxK).
// ---------------------------------------------------------------------------
__device__ __forceinline__ v8f wmma_f16(v16h a, v16h b, v8f c) {
  return __builtin_amdgcn_wmma_f32_16x16x32_f16(
      false, a, false, b, (short)0, c, false, false);
}

__device__ __forceinline__ int lane_kb() { return ((threadIdx.x & 31) >> 4) << 3; }

// fragment from a K-contiguous row pointer (already offset to row + k0 + kb)
__device__ __forceinline__ v16h frag_rm(const half_t* p) {
  v8h lo = *(const v8h*)p;
  v8h hi = *(const v8h*)(p + 16);
  v16h r;
#pragma unroll
  for (int i = 0; i < 8; ++i) { r[i] = lo[i]; r[i + 8] = hi[i]; }
  return r;
}

// A fragment, row-major f16 matrix
__device__ __forceinline__ v16h frag_a(const half_t* A, int lda, int m0, int k0) {
  int lane = threadIdx.x & 31;
  return frag_rm(A + (size_t)(m0 + (lane & 15)) * lda + k0 + lane_kb());
}

// B fragment from TRANSPOSED weights (N x Kpad row-major)
__device__ __forceinline__ v16h frag_bt(const half_t* BT_, int ldk, int n0, int k0) {
  int lane = threadIdx.x & 31;
  return frag_rm(BT_ + (size_t)(n0 + (lane & 15)) * ldk + k0 + lane_kb());
}

// convert 8 contiguous f32 into fragment slots [base, base+8)
__device__ __forceinline__ void cvt8(const float* p, v16h& r, int base) {
  f32x4 a = *(const f32x4*)p;
  f32x4 b = *(const f32x4*)(p + 4);
#pragma unroll
  for (int i = 0; i < 4; ++i) { r[base + i] = (half_t)a[i]; r[base + 4 + i] = (half_t)b[i]; }
}

// ---------------------------------------------------------------------------
// K0: transpose-convert weights: dst[n*Kpad + k] = (f16)src[k*N + n] (0-pad k>=K)
// ---------------------------------------------------------------------------
__global__ void k_cvt_t(const float* __restrict__ src, half_t* __restrict__ dst,
                        int K, int N, int Kpad) {
  int i = blockIdx.x * blockDim.x + threadIdx.x;
  if (i < N * Kpad) {
    int n = i / Kpad, k = i - n * Kpad;
    dst[i] = (k < K) ? (half_t)src[(size_t)k * N + n] : (half_t)0.f;
  }
}

// ---------------------------------------------------------------------------
// K1: xg = X_obs @ Wx + b_gru   [4096 x 768], K=1024
// X_obs(row,k) = obs[row, a=k>>6, f=k&63] (f32, 8-element chunks never straddle
// an agent boundary since chunk starts are 8-aligned).
// Wave computes 16x64 (4 n-tiles, A-fragment reused 4x). 3072 waves.
// ---------------------------------------------------------------------------
__device__ __forceinline__ v16h frag_a_obs(const float* obs, int m0, int k0) {
  int lane = threadIdx.x & 31;
  int row = m0 + (lane & 15);
  int kb = k0 + lane_kb();
  v16h r;
#pragma unroll
  for (int c = 0; c < 2; ++c) {
    int ks = kb + c * 16;
    cvt8(obs + ((size_t)row * A_ + (ks >> 6)) * OD + (ks & 63), r, c * 8);
  }
  return r;
}

__global__ void k_xg(const float* __restrict__ obs, const half_t* __restrict__ WxT,
                     const float* __restrict__ bgru, float* __restrict__ xg) {
  int wave = (blockIdx.x * blockDim.x + threadIdx.x) >> 5;
  int mt = wave / 12, ng = wave % 12;
  int m0 = mt * 16, n0 = ng * 64;
  v8f acc[4] = {};
  for (int k0 = 0; k0 < A_ * F_; k0 += 32) {
    v16h a = frag_a_obs(obs, m0, k0);
#pragma unroll
    for (int j = 0; j < 4; ++j) {
      v16h b = frag_bt(WxT, A_ * F_, n0 + j * 16, k0);
      acc[j] = wmma_f16(a, b, acc[j]);
    }
  }
  int lane = threadIdx.x & 31;
  int mb = m0 + ((lane >> 4) << 3);
#pragma unroll
  for (int j = 0; j < 4; ++j) {
    int n = n0 + j * 16 + (lane & 15);
    float bias = bgru[n];
#pragma unroll
    for (int r = 0; r < 8; ++r)
      xg[(size_t)(mb + r) * (3 * R_) + n] = acc[j][r] + bias;
  }
}

// ---------------------------------------------------------------------------
// K2: sequential GRU (one persistent block, 512 thr = 16 waves), h in LDS.
// hg = h@Wh (wave: 1 m-tile x 6 n-tiles, A reuse 6x) -> gates -> h_new ->
// Hm = sigmoid(h@W_inc + b) written straight to graphs output.
// Dynamic LDS: 32KB h_f32 + 16KB h_f16 + 96KB hg_f32 = 144KB.
// ---------------------------------------------------------------------------
__global__ void k_gru(const float* __restrict__ xg, const half_t* __restrict__ WhT,
                      const half_t* __restrict__ WiT, const float* __restrict__ binc,
                      float* __restrict__ graphs) {
  extern __shared__ char smem[];
  float*  sh_h  = (float*)smem;                     // [32][256]
  half_t* sh_hh = (half_t*)(smem + 32 * 1024);      // [32][256]
  float*  sh_hg = (float*)(smem + 48 * 1024);       // [32][768]

  int tid = threadIdx.x, wid = tid >> 5, lane = tid & 31;

  for (int i = tid; i < BS * R_; i += blockDim.x) { sh_h[i] = 0.f; sh_hh[i] = (half_t)0.f; }
  __syncthreads();

  for (int t = 0; t < T_; ++t) {
    // hg = h @ Wh : [32 x 768], K=256. wave -> m0 = (wid&1)*16, 6 n-tiles
    {
      int m0 = (wid & 1) * 16;
      int nbase = (wid >> 1) * 96;
      v8f acc[6] = {};
      for (int k0 = 0; k0 < R_; k0 += 32) {
        v16h a = frag_a(sh_hh, R_, m0, k0);
#pragma unroll
        for (int j = 0; j < 6; ++j) {
          v16h b = frag_bt(WhT, R_, nbase + j * 16, k0);
          acc[j] = wmma_f16(a, b, acc[j]);
        }
      }
      int mb = m0 + ((lane >> 4) << 3);
#pragma unroll
      for (int j = 0; j < 6; ++j) {
        int n = nbase + j * 16 + (lane & 15);
#pragma unroll
        for (int r = 0; r < 8; ++r) sh_hg[(mb + r) * (3 * R_) + n] = acc[j][r];
      }
    }
    __syncthreads();

    // gates (f32)
    for (int i = tid; i < BS * R_; i += blockDim.x) {
      int b = i >> 8, j = i & 255;
      size_t xrow = ((size_t)b * T_ + t) * (3 * R_);
      float xr = xg[xrow + j],          hr = sh_hg[b * 768 + j];
      float xz = xg[xrow + R_ + j],     hz = sh_hg[b * 768 + R_ + j];
      float xn = xg[xrow + 2 * R_ + j], hn = sh_hg[b * 768 + 2 * R_ + j];
      float rg = 1.f / (1.f + __expf(-(xr + hr)));
      float zg = 1.f / (1.f + __expf(-(xz + hz)));
      float ng = tanhf(xn + rg * hn);
      float hnew = (1.f - zg) * ng + zg * sh_h[i];
      sh_h[i] = hnew;
      sh_hh[i] = (half_t)hnew;
      if (t + 1 < T_) __builtin_prefetch(&xg[xrow + (3 * R_) + j], 0, 1);
    }
    __syncthreads();

    // Hm = sigmoid(h_new @ W_inc + b_inc) : [32 x 128], K=256; 16 tiles/16 waves
    {
      int m0 = (wid & 1) * 16, n0 = (wid >> 1) * 16;
      v8f acc = {};
      for (int k0 = 0; k0 < R_; k0 += 32) {
        v16h a = frag_a(sh_hh, R_, m0, k0);
        v16h b = frag_bt(WiT, R_, n0, k0);
        acc = wmma_f16(a, b, acc);
      }
      int n = n0 + (lane & 15);
      float bias = binc[n];
      int mb = m0 + ((lane >> 4) << 3);
#pragma unroll
      for (int r = 0; r < 8; ++r) {
        float v = 1.f / (1.f + __expf(-(acc[r] + bias)));
        graphs[((size_t)(mb + r) * T_ + t) * (A_ * E_) + n] = v;
      }
    }
    __syncthreads();
  }
}

// ---------------------------------------------------------------------------
// K3: hypergraph conv + emb per (b,t). 128 thr/block, 4096 blocks.
// Tiny einsums in VALU; emb = relu(out @ W_hg) via WMMA (16x64, K=64).
// ---------------------------------------------------------------------------
__global__ void k_hyper(const float* __restrict__ obs, const float* __restrict__ graphs,
                        const half_t* __restrict__ WhgT, half_t* __restrict__ emb) {
  int bt = blockIdx.x;
  int tid = threadIdx.x;
  __shared__ float Hm[A_][E_];
  __shared__ float dvr[A_];
  __shared__ float der[E_];
  __shared__ float xn[A_][F_];
  __shared__ float msg[E_][F_];
  __shared__ __align__(16) half_t outv[A_][F_];

  Hm[tid >> 3][tid & 7] = graphs[(size_t)bt * (A_ * E_) + tid];
  __syncthreads();

  if (tid < A_) {
    float s = EPS;
#pragma unroll
    for (int e = 0; e < E_; ++e) s += Hm[tid][e];
    dvr[tid] = rsqrtf(s);
  } else if (tid < A_ + E_) {
    int e = tid - A_;
    float s = EPS;
#pragma unroll
    for (int a = 0; a < A_; ++a) s += Hm[a][e];
    der[e] = 1.f / s;
  }
  __syncthreads();

  for (int i = tid; i < A_ * F_; i += 128) {
    int a = i >> 6, f = i & 63;
    xn[a][f] = obs[((size_t)bt * A_ + a) * OD + f] * dvr[a];
  }
  __syncthreads();

  for (int i = tid; i < E_ * F_; i += 128) {
    int e = i >> 6, f = i & 63;
    float s = 0.f;
#pragma unroll
    for (int a = 0; a < A_; ++a) s += Hm[a][e] * xn[a][f];
    msg[e][f] = s * der[e];
  }
  __syncthreads();

  for (int i = tid; i < A_ * F_; i += 128) {
    int a = i >> 6, f = i & 63;
    float s = 0.f;
#pragma unroll
    for (int e = 0; e < E_; ++e) s += Hm[a][e] * msg[e][f];
    outv[a][f] = (half_t)(s * dvr[a]);
  }
  __syncthreads();

  int wid = tid >> 5, lane = tid & 31;
  v8f acc = {};
  for (int k0 = 0; k0 < F_; k0 += 32) {
    v16h a = frag_a(&outv[0][0], F_, 0, k0);
    v16h b = frag_bt(WhgT, F_, wid * 16, k0);
    acc = wmma_f16(a, b, acc);
  }
  int n = wid * 16 + (lane & 15);
  int mb = (lane >> 4) << 3;
#pragma unroll
  for (int r = 0; r < 8; ++r) {
    float v = fmaxf(acc[r], 0.f);
    emb[((size_t)bt * A_ + (mb + r)) * EMB + n] = (half_t)v;
  }
}

// ---------------------------------------------------------------------------
// K4: H1 = relu(X1 @ W1 + b1)  [65536 x 256], K=352 padded.
// X1 chunks (8-aligned) never straddle segment boundaries (256/272/336).
// Wave computes 16x64 (4 n-tiles). 16384 waves.
// ---------------------------------------------------------------------------
__device__ __forceinline__ void mlp_chunk(const float* state, const half_t* emb,
                                          int row, int ks, v16h& r, int base) {
  if (ks < SD) {
    cvt8(state + (size_t)(row >> 4) * SD + ks, r, base);
  } else if (ks < SD + A_) {
    int a = row & 15;
#pragma unroll
    for (int j = 0; j < 8; ++j) r[base + j] = (half_t)((a == ks - SD + j) ? 1.f : 0.f);
  } else if (ks < SD + A_ + EMB) {
    v8h v = *(const v8h*)(emb + (size_t)row * EMB + (ks - SD - A_));
#pragma unroll
    for (int j = 0; j < 8; ++j) r[base + j] = v[j];
  } else {
#pragma unroll
    for (int j = 0; j < 8; ++j) r[base + j] = (half_t)0.f;
  }
}

__device__ __forceinline__ v16h frag_a_mlp(const float* state, const half_t* emb,
                                           int m0, int k0) {
  int lane = threadIdx.x & 31;
  int row = m0 + (lane & 15);
  int kb = k0 + lane_kb();
  v16h r;
  mlp_chunk(state, emb, row, kb, r, 0);
  mlp_chunk(state, emb, row, kb + 16, r, 8);
  return r;
}

__global__ void k_mlp1(const float* __restrict__ state, const half_t* __restrict__ emb,
                       const half_t* __restrict__ W1T, const float* __restrict__ b1,
                       half_t* __restrict__ H1) {
  int wave = (blockIdx.x * blockDim.x + threadIdx.x) >> 5;
  int m0 = (wave >> 2) * 16, n0 = (wave & 3) * 64;
  v8f acc[4] = {};
  for (int k0 = 0; k0 < K1PAD; k0 += 32) {
    v16h a = frag_a_mlp(state, emb, m0, k0);
#pragma unroll
    for (int j = 0; j < 4; ++j) {
      v16h b = frag_bt(W1T, K1PAD, n0 + j * 16, k0);
      acc[j] = wmma_f16(a, b, acc[j]);
    }
  }
  int lane = threadIdx.x & 31;
  int mb = m0 + ((lane >> 4) << 3);
#pragma unroll
  for (int j = 0; j < 4; ++j) {
    int n = n0 + j * 16 + (lane & 15);
    float bias = b1[n];
#pragma unroll
    for (int r = 0; r < 8; ++r) {
      float v = fmaxf(acc[j][r] + bias, 0.f);
      H1[(size_t)(mb + r) * HID + n] = (half_t)v;
    }
  }
}

// ---------------------------------------------------------------------------
// K5: H2 = relu(H1 @ W2 + b2); q = H2 @ W3 + b3 (fused N=1 epilogue).
// Block per 16-row tile; A-tile (16x256 f16, 8KB) staged into LDS by the
// Tensor Data Mover (tensor_load_to_lds, 2D descriptor, data_size=2B,
// tile 256x16, stride 256), then s_wait_tensorcnt + barrier; A-fragments
// come from LDS (ds_load_b128). Falls back to cooperative copy if the TDM
// builtin is unavailable.
// ---------------------------------------------------------------------------
__global__ void k_mlp23(const half_t* __restrict__ H1, const half_t* __restrict__ W2T,
                        const float* __restrict__ b2, const float* __restrict__ W3,
                        const float* __restrict__ b3, float* __restrict__ q) {
  int m0 = blockIdx.x * 16;
  int tid = threadIdx.x, wid = tid >> 5, lane = tid & 31;

  // single static LDS block => shA sits at LDS offset 0 (used as TDM lds_addr)
  __shared__ __align__(16) unsigned char smem_blk[16 * HID * 2 + 16 * HID * 4];
  half_t* shA = (half_t*)smem_blk;                 // [16][256] f16  (offset 0)
  float*  sh2 = (float*)(smem_blk + 16 * HID * 2); // [16][256] f32

#if USE_TDM
  if (tid == 0) {
    unsigned long long ga = (unsigned long long)(uintptr_t)(H1 + (size_t)m0 * HID);
    u32x4 g0;
    g0[0] = 1u;                                   // count=1, no gather
    g0[1] = 0u;                                   // lds_addr = 0 (start of block)
    g0[2] = (unsigned int)ga;                     // global_addr[31:0]
    g0[3] = ((unsigned int)(ga >> 32) & 0x01FFFFFFu) | 0x80000000u; // addr[56:32] | type=2
    i32x8 g1;
    g1[0] = 1 << 16;                              // data_size = 2 bytes
    g1[1] = (int)((HID & 0xFFFF) << 16);          // tensor_dim0 = 256 (low 16)
    g1[2] = (int)((HID >> 16) | ((NROWS & 0xFFFF) << 16)); // dim0 hi | dim1 lo
    g1[3] = (int)((NROWS >> 16) | (HID << 16));   // dim1 hi | tile_dim0 = 256
    g1[4] = 16;                                   // tile_dim1 = 16, tile_dim2 = 0
    g1[5] = HID;                                  // tensor_dim0_stride = 256
    g1[6] = 0;
    g1[7] = 0;
#if defined(__clang_major__) && __clang_major__ >= 23
    __builtin_amdgcn_tensor_load_to_lds(g0, g1, (i32x4)0, (i32x4)0, (i32x8)0, 0);
#else
    __builtin_amdgcn_tensor_load_to_lds(g0, g1, (i32x4)0, (i32x4)0, 0);
#endif
  }
  __builtin_amdgcn_s_wait_tensorcnt(0);
  __syncthreads();
#else
  for (int c = tid; c < 16 * HID / 8; c += blockDim.x) {
    *(v8h*)(shA + c * 8) = *(const v8h*)(H1 + (size_t)m0 * HID + c * 8);
  }
  __syncthreads();
#endif

  // H2 tile: wave handles n-tiles wid and wid+8 with A-fragment reuse
  {
    v8f acc0 = {}, acc1 = {};
    int n0 = wid * 16, n1 = (wid + 8) * 16;
    for (int k0 = 0; k0 < HID; k0 += 32) {
      v16h a = frag_a(shA, HID, 0, k0);           // from LDS
      v16h b0 = frag_bt(W2T, HID, n0, k0);
      v16h b1 = frag_bt(W2T, HID, n1, k0);
      acc0 = wmma_f16(a, b0, acc0);
      acc1 = wmma_f16(a, b1, acc1);
    }
    int mb = (lane >> 4) << 3;
    int nn0 = n0 + (lane & 15), nn1 = n1 + (lane & 15);
    float bi0 = b2[nn0], bi1 = b2[nn1];
#pragma unroll
    for (int r = 0; r < 8; ++r) {
      sh2[(mb + r) * HID + nn0] = fmaxf(acc0[r] + bi0, 0.f);
      sh2[(mb + r) * HID + nn1] = fmaxf(acc1[r] + bi1, 0.f);
    }
  }
  __syncthreads();

  if (tid < 16) {
    float s = b3[0];
    for (int c = 0; c < HID; ++c) s += sh2[tid * HID + c] * W3[c];
    q[m0 + tid] = s;
  }
}

// ---------------------------------------------------------------------------
// Host launcher
// ---------------------------------------------------------------------------
extern "C" void kernel_launch(void* const* d_in, const int* in_sizes, int n_in,
                              void* d_out, int out_size, void* d_ws, size_t ws_size,
                              hipStream_t stream) {
  const float* state = (const float*)d_in[0];
  const float* obs   = (const float*)d_in[1];
  const float* Wx    = (const float*)d_in[2];
  const float* Wh    = (const float*)d_in[3];
  const float* bgru  = (const float*)d_in[4];
  const float* Winc  = (const float*)d_in[5];
  const float* binc  = (const float*)d_in[6];
  const float* Whg   = (const float*)d_in[7];
  const float* W1    = (const float*)d_in[8];
  const float* b1    = (const float*)d_in[9];
  const float* W2    = (const float*)d_in[10];
  const float* b2    = (const float*)d_in[11];
  const float* W3    = (const float*)d_in[12];
  const float* b3    = (const float*)d_in[13];

  float* q      = (float*)d_out;               // [32,128,16,1]
  float* graphs = q + NROWS;                   // [32,128,16,8]

  // Workspace layout (bytes); all transposed weights are N x Kpad f16
  char* ws = (char*)d_ws;
  half_t* WxT  = (half_t*)(ws + 0);            // 768 x 1024
  half_t* WhT  = (half_t*)(ws + 1572864);      // 768 x 256
  half_t* WiT  = (half_t*)(ws + 1966080);      // 128 x 256
  half_t* W1T  = (half_t*)(ws + 2031616);      // 256 x 352 (0-padded K)
  half_t* W2T  = (half_t*)(ws + 2211840);      // 256 x 256
  half_t* WhgT = (half_t*)(ws + 2342912);      //  64 x 64
  float*  xg   = (float*) (ws + 2351104);      // 4096 x 768 f32
  half_t* emb  = (half_t*)(ws + 14934016);     // 65536 x 64 f16
  half_t* H1   = (half_t*)(ws + 23322624);     // 65536 x 256 f16
  (void)in_sizes; (void)n_in; (void)out_size; (void)ws_size;

  // K0: transpose-convert weights (src K x N -> dst N x Kpad, f16)
  k_cvt_t<<<(768 * 1024 + 255) / 256, 256, 0, stream>>>(Wx,  WxT,  1024, 768, 1024);
  k_cvt_t<<<(768 * 256 + 255) / 256, 256, 0, stream>>>(Wh,  WhT,  256, 768, 256);
  k_cvt_t<<<(128 * 256 + 255) / 256, 256, 0, stream>>>(Winc, WiT, 256, 128, 256);
  k_cvt_t<<<(256 * 352 + 255) / 256, 256, 0, stream>>>(W1,  W1T,  336, 256, 352);
  k_cvt_t<<<(256 * 256 + 255) / 256, 256, 0, stream>>>(W2,  W2T,  256, 256, 256);
  k_cvt_t<<<(64 * 64 + 255) / 256, 256, 0, stream>>>(Whg, WhgT, 64, 64, 64);

  // K1: xg (3072 waves, 8 per block)
  k_xg<<<384, 256, 0, stream>>>(obs, WxT, bgru, xg);

  // K2: sequential GRU + incidence -> graphs (one block, 144KB dyn LDS)
  k_gru<<<1, 512, 147456, stream>>>(xg, WhT, WiT, binc, graphs);

  // K3: hypergraph conv + emb
  k_hyper<<<BT, 128, 0, stream>>>(obs, graphs, WhgT, emb);

  // K4: MLP layer 1 (16384 waves)
  k_mlp1<<<2048, 256, 0, stream>>>(state, emb, W1T, b1, H1);

  // K5: MLP layers 2+3 fused (TDM-staged A tile)
  k_mlp23<<<NROWS / 16, 256, 0, stream>>>(H1, W2T, b2, W3, b3, q);
}
